// TimeMixing_18648747999321
// MI455X (gfx1250) — matrix operand order repaired and checked by reading, not compiled, not run on hardware
//
#include <hip/hip_runtime.h>
#include <hip/hip_bf16.h>
#include <stdint.h>

// ---------------------------------------------------------------------------
// RWKV time-mixing for MI455X (gfx1250, wave32, WMMA).
//   B=8, T=2048, C=1024.  GEMMs run as bf16 V_WMMA_F32_16X16X32_BF16 with
//   f32 accumulation (64x32 tile per wave, 8 WMMA per K-step, pure
//   global_load_b128 fragment feeds); scan + elementwise run as f32 VALU.
// ---------------------------------------------------------------------------

typedef unsigned short ushort_t;
typedef __bf16 bf16_t;
typedef bf16_t v16bf __attribute__((ext_vector_type(16)));
typedef bf16_t v8bf  __attribute__((ext_vector_type(8)));
typedef float  v8f   __attribute__((ext_vector_type(8)));

// Explicit global-address-space pointers: keep fragment loads on the
// global_load path (LOADcnt only), never generic/flat (LOADcnt+DScnt).
typedef const __attribute__((address_space(1))) ushort_t* gushort_p;
typedef const __attribute__((address_space(1))) v8bf*     gv8bf_p;

union FragAB { v16bf v; v8bf h[2]; };

__device__ __forceinline__ ushort_t f2bf_bits(float f) {
  unsigned u = __builtin_bit_cast(unsigned, f);
  // round-to-nearest-even
  u = (u + 0x7FFFu + ((u >> 16) & 1u)) >> 16;
  return (ushort_t)u;
}

__device__ __forceinline__ gushort_p as_global(const ushort_t* p) {
  return (gushort_p)(uintptr_t)p;
}

__device__ __forceinline__ v8bf load8bf_g(gushort_p p) {
  return *(gv8bf_p)p;
}

// ------------------------------ f32 -> bf16 --------------------------------
__global__ void cvt_bf16_kernel(const float* __restrict__ src,
                                ushort_t* __restrict__ dst, int n) {
  int i = blockIdx.x * blockDim.x + threadIdx.x;
  if (i < n) dst[i] = f2bf_bits(src[i]);
}

// --------------------- time-shift + token mixing (bf16 out) ----------------
__global__ void mix_kernel(const float* __restrict__ x,
                           const float* __restrict__ mk,
                           const float* __restrict__ mv,
                           const float* __restrict__ mr,
                           ushort_t* __restrict__ kin,
                           ushort_t* __restrict__ vin,
                           ushort_t* __restrict__ rin,
                           int T, int C, long long n) {
  long long i = (long long)blockIdx.x * blockDim.x + threadIdx.x;
  if (i >= n) return;
  int c = (int)(i % C);
  long long bt = i / C;
  int t = (int)(bt % T);
  float xv = x[i];
  float px = (t == 0) ? 0.0f : x[i - C];
  float a = mk[c], b = mv[c], r = mr[c];
  kin[i] = f2bf_bits(xv * a + px * (1.0f - a));
  vin[i] = f2bf_bits(xv * b + px * (1.0f - b));
  rin[i] = f2bf_bits(xv * r + px * (1.0f - r));
}

// --------------------------- bf16 WMMA GEMM --------------------------------
// Out[m][n] = sum_k A[m][k] * W[n][k]   (A: MxK row-major, W: NxK row-major)
// Block = 256 threads = 8 waves, waves tiled 2(M) x 4(N); each wave computes
// a 64x32 tile (4x2 grid of 16x16x32 WMMA tiles, K-loop step 32):
//   8 WMMA per K-step against 12 global_load_b128 (B-frags reused 4x).
__global__ void __launch_bounds__(256)
gemm_bf16_kernel(const ushort_t* __restrict__ A,
                 const ushort_t* __restrict__ W,
                 float* __restrict__ Out,
                 int M, int N, int K) {
  const int lane = threadIdx.x & 31;
  const int wid  = threadIdx.x >> 5;
  const int r    = lane & 15;       // row within half-wave
  const int half = lane >> 4;       // 0: lanes 0-15, 1: lanes 16-31

  const int m0 = blockIdx.x * 128 + (wid & 1) * 64;   // wave M origin
  const int n0 = blockIdx.y * 128 + (wid >> 1) * 32;  // wave N origin

  // A-fragment 16x32 bf16: lanes<16 hold K {0..7,16..23}, lanes>=16 {8..15,24..31}
  const int aoff = half * 8;
  // B-fragment 32x16 bf16: lanes<16 hold K {0..7,16..23}, lanes>=16 {8..15,24..31}
  const int boff = half * 8;

  // Per-fragment global-AS base pointers (indexed by k0 so the backend can
  // strength-reduce to base + immediate-offset global_load_b128 forms).
  gushort_p pa0 = as_global(A + (size_t)(m0 +  0 + r) * K + aoff);
  gushort_p pa1 = as_global(A + (size_t)(m0 + 16 + r) * K + aoff);
  gushort_p pa2 = as_global(A + (size_t)(m0 + 32 + r) * K + aoff);
  gushort_p pa3 = as_global(A + (size_t)(m0 + 48 + r) * K + aoff);
  gushort_p pb0 = as_global(W + (size_t)(n0 +  0 + r) * K + boff);
  gushort_p pb1 = as_global(W + (size_t)(n0 + 16 + r) * K + boff);

  const ushort_t* pf0 = A + (size_t)(m0 +  0 + r) * K + aoff;
  const ushort_t* pf2 = A + (size_t)(m0 + 32 + r) * K + aoff;

  v8f acc00 = {}, acc01 = {}, acc10 = {}, acc11 = {};
  v8f acc20 = {}, acc21 = {}, acc30 = {}, acc31 = {};

  for (int k0 = 0; k0 < K; k0 += 32) {
    if (k0 + 256 < K) {
      __builtin_prefetch(pf0 + k0 + 256, 0, 1);
      __builtin_prefetch(pf2 + k0 + 256, 0, 1);
    }
    FragAB b0, b1;
    b0.h[0] = load8bf_g(pb0 + k0);
    b0.h[1] = load8bf_g(pb0 + k0 + 16);
    b1.h[0] = load8bf_g(pb1 + k0);
    b1.h[1] = load8bf_g(pb1 + k0 + 16);

    FragAB a;
    a.h[0] = load8bf_g(pa0 + k0);
    a.h[1] = load8bf_g(pa0 + k0 + 16);
    acc00 = __builtin_amdgcn_wmma_f32_16x16x32_bf16(
        false, a.v, false, b0.v, (short)0, acc00, false, false);
    acc01 = __builtin_amdgcn_wmma_f32_16x16x32_bf16(
        false, a.v, false, b1.v, (short)0, acc01, false, false);

    a.h[0] = load8bf_g(pa1 + k0);
    a.h[1] = load8bf_g(pa1 + k0 + 16);
    acc10 = __builtin_amdgcn_wmma_f32_16x16x32_bf16(
        false, a.v, false, b0.v, (short)0, acc10, false, false);
    acc11 = __builtin_amdgcn_wmma_f32_16x16x32_bf16(
        false, a.v, false, b1.v, (short)0, acc11, false, false);

    a.h[0] = load8bf_g(pa2 + k0);
    a.h[1] = load8bf_g(pa2 + k0 + 16);
    acc20 = __builtin_amdgcn_wmma_f32_16x16x32_bf16(
        false, a.v, false, b0.v, (short)0, acc20, false, false);
    acc21 = __builtin_amdgcn_wmma_f32_16x16x32_bf16(
        false, a.v, false, b1.v, (short)0, acc21, false, false);

    a.h[0] = load8bf_g(pa3 + k0);
    a.h[1] = load8bf_g(pa3 + k0 + 16);
    acc30 = __builtin_amdgcn_wmma_f32_16x16x32_bf16(
        false, a.v, false, b0.v, (short)0, acc30, false, false);
    acc31 = __builtin_amdgcn_wmma_f32_16x16x32_bf16(
        false, a.v, false, b1.v, (short)0, acc31, false, false);
  }

  // C/D layout: VGPR e -> row (e + 8*half), column = lane&15.
  v8f acc[4][2] = {{acc00, acc01}, {acc10, acc11}, {acc20, acc21}, {acc30, acc31}};
#pragma unroll
  for (int i = 0; i < 4; ++i) {
    float* Orow = Out + (size_t)(m0 + 16 * i + 8 * half) * N + n0 + r;
#pragma unroll
    for (int e = 0; e < 8; ++e) {
      Orow[(size_t)e * N]      = acc[i][0][e];
      Orow[(size_t)e * N + 16] = acc[i][1][e];
    }
  }
}

// ------------------ WKV scan + sigmoid(r) * wkv (bf16 out) -----------------
// Threads cover channel dim (coalesced per timestep), sequential over T.
// Critical path is the dependent state chain; use fast rcp-based division.
__global__ void __launch_bounds__(256)
wkv_scan_kernel(const float* __restrict__ key,
                const float* __restrict__ value,
                const float* __restrict__ rlog,
                const float* __restrict__ time_decay,
                const float* __restrict__ time_first,
                ushort_t* __restrict__ rwkv,
                int T, int C) {
  int c = blockIdx.x * blockDim.x + threadIdx.x;
  int b = blockIdx.y;
  if (c >= C) return;

  const float w = -__expf(time_decay[c]);
  const float u = time_first[c];

  float num = 0.0f, den = 0.0f, mx = -1e38f;
  size_t base = ((size_t)b * T) * C + c;

  for (int t = 0; t < T; ++t) {
    size_t idx = base + (size_t)t * C;
    float k = key[idx];
    float v = value[idx];

    float mo  = fmaxf(mx, k + u);
    float e1  = __expf(mx - mo);
    float e2  = __expf(k + u - mo);
    float out = __fdividef(e1 * num + e2 * v, e1 * den + e2);

    float ms  = fmaxf(mx + w, k);
    float e1s = __expf(mx + w - ms);
    float e2s = __expf(k - ms);
    num = e1s * num + e2s * v;
    den = e1s * den + e2s;
    mx  = ms;

    float rl = rlog[idx];
    float rr = __fdividef(1.0f, 1.0f + __expf(-rl));
    rwkv[idx] = f2bf_bits(rr * out);
  }
}

// ---------------------------------------------------------------------------
extern "C" void kernel_launch(void* const* d_in, const int* in_sizes, int n_in,
                              void* d_out, int out_size, void* d_ws, size_t ws_size,
                              hipStream_t stream) {
  (void)n_in; (void)ws_size; (void)out_size;

  const float* x   = (const float*)d_in[0];
  const float* wk  = (const float*)d_in[1];
  const float* wv  = (const float*)d_in[2];
  const float* wr  = (const float*)d_in[3];
  const float* wo  = (const float*)d_in[4];
  const float* td  = (const float*)d_in[5];
  const float* tf  = (const float*)d_in[6];
  const float* mk  = (const float*)d_in[7];
  const float* mv  = (const float*)d_in[8];
  const float* mr  = (const float*)d_in[9];

  const int C  = in_sizes[5];           // 1024
  const int T  = 2048;
  const long long BTC = in_sizes[0];    // B*T*C
  const int B  = (int)(BTC / ((long long)T * C));
  const int M  = B * T;                 // 16384

  // --- carve workspace (all 256B aligned; sizes are powers of two) ---
  char* ws = (char*)d_ws;
  size_t off = 0;
  auto carve = [&](size_t bytes) {
    char* p = ws + off;
    off += (bytes + 255) & ~(size_t)255;
    return p;
  };
  ushort_t* kin  = (ushort_t*)carve((size_t)BTC * 2);
  ushort_t* vin  = (ushort_t*)carve((size_t)BTC * 2);
  ushort_t* rin  = (ushort_t*)carve((size_t)BTC * 2);
  ushort_t* rwkv = (ushort_t*)carve((size_t)BTC * 2);
  ushort_t* wkb  = (ushort_t*)carve((size_t)C * C * 2);
  ushort_t* wvb  = (ushort_t*)carve((size_t)C * C * 2);
  ushort_t* wrb  = (ushort_t*)carve((size_t)C * C * 2);
  ushort_t* wob  = (ushort_t*)carve((size_t)C * C * 2);
  float*    keyf = (float*)carve((size_t)BTC * 4);
  float*    valf = (float*)carve((size_t)BTC * 4);
  float*    rlog = (float*)carve((size_t)BTC * 4);

  // 1) weights f32 -> bf16 (2 MB each as bf16; resident in 192 MB L2)
  {
    int n = C * C;
    int blocks = (n + 255) / 256;
    cvt_bf16_kernel<<<blocks, 256, 0, stream>>>(wk, wkb, n);
    cvt_bf16_kernel<<<blocks, 256, 0, stream>>>(wv, wvb, n);
    cvt_bf16_kernel<<<blocks, 256, 0, stream>>>(wr, wrb, n);
    cvt_bf16_kernel<<<blocks, 256, 0, stream>>>(wo, wob, n);
  }

  // 2) fused time-shift + mix -> bf16 GEMM inputs
  {
    long long n = BTC;
    int blocks = (int)((n + 255) / 256);
    mix_kernel<<<blocks, 256, 0, stream>>>(x, mk, mv, mr, kin, vin, rin, T, C, n);
  }

  // 3) three WMMA GEMMs: key / value / receptance-logit
  {
    dim3 grid(M / 128, C / 128);
    gemm_bf16_kernel<<<grid, 256, 0, stream>>>(kin, wkb, keyf, M, C, C);
    gemm_bf16_kernel<<<grid, 256, 0, stream>>>(vin, wvb, valf, M, C, C);
    gemm_bf16_kernel<<<grid, 256, 0, stream>>>(rin, wrb, rlog, M, C, C);
  }

  // 4) sequential WKV scan fused with sigmoid(r)*wkv, bf16 output
  {
    dim3 grid(C / 256, B);
    wkv_scan_kernel<<<grid, 256, 0, stream>>>(keyf, valf, rlog, td, tf, rwkv, T, C);
  }

  // 5) output projection: out = rwkv @ wo^T  (f32 result)
  {
    dim3 grid(M / 128, C / 128);
    gemm_bf16_kernel<<<grid, 256, 0, stream>>>(rwkv, wob, (float*)d_out, M, C, C);
  }
}
